// Wavenet2D_38551626449098
// MI455X (gfx1250) — matrix operand order repaired and checked by reading, not compiled, not run on hardware
//
#include <hip/hip_runtime.h>
#include <hip/hip_bf16.h>

// ---------------------------------------------------------------------------
// WaveNet-2D on MI455X (gfx1250): implicit-GEMM via v_wmma_f32_16x16x32_bf16.
// B=2, H=16, W=4096, C=128, 2C=256, cond=80 (padded to 128), 8 layers.
// Compute-bound (~7.3e11 FLOPs vs ~0.5GB traffic) -> bf16 WMMA, f32 accum.
// ---------------------------------------------------------------------------

typedef __attribute__((ext_vector_type(16))) __bf16 v16bf;
typedef __attribute__((ext_vector_type(8)))  float  v8f;

struct alignas(16) Q16 { unsigned long long lo, hi; };
union Frag { v16bf v; Q16 q[2]; };

#define BB   2
#define HH   16
#define WW   4096
#define CC   128
#define C2   256
#define HW   (HH * WW)            // 65536
#define NTOT (BB * HW)            // 131072 spatial positions
#define LP   136                  // LDS pitch in bf16 elems (128 + 8 pad)
#define NL   8

__device__ __forceinline__ unsigned short f2bf(float f) {
    unsigned u = __float_as_uint(f);
    u += 0x7FFFu + ((u >> 16) & 1u);          // round-to-nearest-even
    return (unsigned short)(u >> 16);
}

// ------------------------- weight / input conversion -----------------------

// fg_w [8][256][128][3][3] f32 -> wfg [8][9(tap)][256][128] bf16
__global__ void k_cvt_fgw(const float* __restrict__ src, unsigned short* __restrict__ dst) {
    int idx = blockIdx.x * blockDim.x + threadIdx.x;
    if (idx >= NL * 9 * C2 * CC) return;
    int l  = idx / (9 * C2 * CC);
    int r  = idx % (9 * C2 * CC);
    int t  = r / (C2 * CC);
    int r2 = r % (C2 * CC);
    int m  = r2 >> 7;
    int k  = r2 & 127;
    int kh = t / 3, kw = t % 3;
    float v = src[((((size_t)l * C2 + m) * CC + k) * 3 + kh) * 3 + kw];
    dst[idx] = f2bf(v);
}

// fgc_w [8][256][80] f32 -> wfgc [8][256][128] bf16 (K padded with zeros)
__global__ void k_cvt_fgc(const float* __restrict__ src, unsigned short* __restrict__ dst) {
    int idx = blockIdx.x * blockDim.x + threadIdx.x;
    if (idx >= NL * C2 * CC) return;
    int l = idx >> 15;
    int r = idx & 32767;
    int m = r >> 7;
    int k = r & 127;
    float v = (k < 80) ? src[((size_t)l * C2 + m) * 80 + k] : 0.0f;
    dst[idx] = f2bf(v);
}

// rs_w [8][256][128] f32 -> bf16 (same layout)
__global__ void k_cvt_rs(const float* __restrict__ src, unsigned short* __restrict__ dst) {
    int idx = blockIdx.x * blockDim.x + threadIdx.x;
    if (idx >= NL * C2 * CC) return;
    dst[idx] = f2bf(src[idx]);
}

// c [2][80][16][4096] f32 -> bf16
__global__ void k_cvt_c(const float* __restrict__ src, unsigned short* __restrict__ dst) {
    int idx = blockIdx.x * blockDim.x + threadIdx.x;
    if (idx >= BB * 80 * HW) return;
    dst[idx] = f2bf(src[idx]);
}

// front 1x1 conv from 1 channel: h = fw[c]*x + fb[c]; write f32 + bf16 shadow
__global__ void k_front(const float* __restrict__ x, const float* __restrict__ fw,
                        const float* __restrict__ fb, float* __restrict__ hf,
                        unsigned short* __restrict__ hbf) {
    int idx = blockIdx.x * blockDim.x + threadIdx.x;
    if (idx >= BB * CC * HW) return;
    int b   = idx / (CC * HW);
    int wi  = idx % (CC * HW);
    int cc  = wi >> 16;              // / HW
    int sp  = wi & (HW - 1);
    float v = fw[cc] * x[(size_t)b * HW + sp] + fb[cc];
    hf[idx]  = v;
    hbf[idx] = f2bf(v);
}

// ------------------------------- gate kernel -------------------------------
// Tile = 64 consecutive W positions (row aligned). 8 waves; wave w owns
// filter rows [16w,16w+16) and gate rows [128+16w, 128+16w+16).
// Taps 0..8 = shifted 3x3 conv GEMMs (K=128); tap 9 = cond GEMM (K=128 padded).

__global__ __launch_bounds__(256)
void k_gate(const unsigned short* __restrict__ hbf, const unsigned short* __restrict__ cbf,
            const unsigned short* __restrict__ wfg, const unsigned short* __restrict__ wfgc,
            const float* __restrict__ fg_b, const float* __restrict__ fgc_b,
            unsigned short* __restrict__ acts, int layer, int dh, int dw) {
    __shared__ unsigned short ldsB[64 * LP];

    const int tid  = threadIdx.x;
    const int lane = tid & 31;
    const int wv   = tid >> 5;

    const int n0 = blockIdx.x * 64;
    const int b  = n0 >> 16;           // / HW
    const int rm = n0 & (HW - 1);
    const int hh = rm >> 12;           // / W
    const int w0 = rm & (WW - 1);

    v8f accF[4] = {};
    v8f accG[4] = {};

    const unsigned short* wfgL  = wfg  + (size_t)layer * 9 * C2 * CC;
    const unsigned short* wfgcL = wfgc + (size_t)layer * C2 * CC;
    __builtin_prefetch(wfgL, 0, 1);    // global_prefetch_b8

    const int mF   = wv * 16 + (lane & 15);
    const int aoff = (lane >> 4) * 8;
    const int nc   = lane & 15;

    for (int t = 0; t < 10; ++t) {
        __syncthreads();               // protect LDS vs previous tap's reads
        if (t < 9) {
            const int kh = t / 3, kw = t % 3;
            const int oh = (kh - 2) * dh;     // causal: pad_h = 2*dh on top
            const int ow = (kw - 1) * dw;
            const int hs = hh + oh;
            if (hs < 0) continue;             // whole tap is zero padding (uniform)
            const size_t rowbase = (((size_t)b * CC) * HH + hs) * WW;
            #pragma unroll 4
            for (int j = 0; j < 32; ++j) {
                int elem = tid + (j << 8);
                int pos  = elem & 63;
                int chan = elem >> 6;
                int ws   = w0 + ow + pos;
                unsigned short v = 0;
                if (ws >= 0 && ws < WW)
                    v = hbf[rowbase + (size_t)chan * HW + ws];
                ldsB[pos * LP + chan] = v;
            }
        } else {                               // conditioning "tap"
            const size_t cbase = (((size_t)b * 80) * HH + hh) * WW + w0;
            #pragma unroll 4
            for (int j = 0; j < 32; ++j) {
                int elem = tid + (j << 8);
                int pos  = elem & 63;
                int chan = elem >> 6;
                unsigned short v = 0;
                if (chan < 80)
                    v = cbf[cbase + (size_t)chan * HW + pos];
                ldsB[pos * LP + chan] = v;
            }
        }
        __syncthreads();

        const unsigned short* wb    = (t < 9) ? (wfgL + (size_t)t * C2 * CC) : wfgcL;
        const unsigned short* wrowF = wb + (size_t)mF * CC;
        const unsigned short* wrowG = wb + (size_t)(mF + 128) * CC;
        if (t < 9) __builtin_prefetch(wfgL + (size_t)(t + 1) * C2 * CC, 0, 1);

        #pragma unroll
        for (int kc = 0; kc < 4; ++kc) {
            const int kb = kc * 32;
            Frag af, ag;
            af.q[0] = *(const Q16*)(wrowF + kb + aoff);
            af.q[1] = *(const Q16*)(wrowF + kb + aoff + 16);
            ag.q[0] = *(const Q16*)(wrowG + kb + aoff);
            ag.q[1] = *(const Q16*)(wrowG + kb + aoff + 16);
            #pragma unroll
            for (int ni = 0; ni < 4; ++ni) {
                const unsigned short* bp = &ldsB[(ni * 16 + nc) * LP + kb + ((lane >> 4) << 3)];
                Frag bm;
                bm.q[0] = *(const Q16*)(bp);
                bm.q[1] = *(const Q16*)(bp + 16);
                accF[ni] = __builtin_amdgcn_wmma_f32_16x16x32_bf16(
                    false, af.v, false, bm.v, (short)0, accF[ni], false, false);
                accG[ni] = __builtin_amdgcn_wmma_f32_16x16x32_bf16(
                    false, ag.v, false, bm.v, (short)0, accG[ni], false, false);
            }
        }
    }

    // epilogue: gated activation, wave-local (filter row m pairs with gate m+128)
    const float* fgb  = fg_b  + layer * C2;
    const float* fgcb = fgc_b + layer * C2;
    const int rowsel  = (lane >> 4) << 3;
    #pragma unroll
    for (int ni = 0; ni < 4; ++ni) {
        const int col = ni * 16 + nc;
        #pragma unroll
        for (int v = 0; v < 8; ++v) {
            const int m  = wv * 16 + rowsel + v;
            float af = accF[ni][v] + fgb[m]       + fgcb[m];
            float ag = accG[ni][v] + fgb[m + 128] + fgcb[m + 128];
            float act = tanhf(af) * (1.0f / (1.0f + expf(-ag)));
            acts[(((size_t)b * CC + m) * HH + hh) * WW + w0 + col] = f2bf(act);
        }
    }
}

// ------------------------------ res/skip kernel ----------------------------

__global__ __launch_bounds__(256)
void k_rs(const unsigned short* __restrict__ actsb, const unsigned short* __restrict__ wrs,
          const float* __restrict__ rs_b, float* __restrict__ hf,
          unsigned short* __restrict__ hbf, float* __restrict__ skip,
          int layer, int first) {
    __shared__ unsigned short ldsB[64 * LP];

    const int tid  = threadIdx.x;
    const int lane = tid & 31;
    const int wv   = tid >> 5;

    const int n0 = blockIdx.x * 64;
    const int b  = n0 >> 16;
    const int rm = n0 & (HW - 1);
    const int hh = rm >> 12;
    const int w0 = rm & (WW - 1);

    v8f accR[4] = {};
    v8f accS[4] = {};

    const size_t abase = (((size_t)b * CC) * HH + hh) * WW + w0;
    #pragma unroll 4
    for (int j = 0; j < 32; ++j) {
        int elem = tid + (j << 8);
        int pos  = elem & 63;
        int chan = elem >> 6;
        ldsB[pos * LP + chan] = actsb[abase + (size_t)chan * HW + pos];
    }
    __syncthreads();

    const unsigned short* wb = wrs + (size_t)layer * C2 * CC;
    const int mR   = wv * 16 + (lane & 15);
    const int aoff = (lane >> 4) * 8;
    const int nc   = lane & 15;
    const unsigned short* wrowR = wb + (size_t)mR * CC;
    const unsigned short* wrowS = wb + (size_t)(mR + 128) * CC;
    __builtin_prefetch(wrowR, 0, 1);

    #pragma unroll
    for (int kc = 0; kc < 4; ++kc) {
        const int kb = kc * 32;
        Frag ar, as_;
        ar.q[0]  = *(const Q16*)(wrowR + kb + aoff);
        ar.q[1]  = *(const Q16*)(wrowR + kb + aoff + 16);
        as_.q[0] = *(const Q16*)(wrowS + kb + aoff);
        as_.q[1] = *(const Q16*)(wrowS + kb + aoff + 16);
        #pragma unroll
        for (int ni = 0; ni < 4; ++ni) {
            const unsigned short* bp = &ldsB[(ni * 16 + nc) * LP + kb + ((lane >> 4) << 3)];
            Frag bm;
            bm.q[0] = *(const Q16*)(bp);
            bm.q[1] = *(const Q16*)(bp + 16);
            accR[ni] = __builtin_amdgcn_wmma_f32_16x16x32_bf16(
                false, ar.v, false, bm.v, (short)0, accR[ni], false, false);
            accS[ni] = __builtin_amdgcn_wmma_f32_16x16x32_bf16(
                false, as_.v, false, bm.v, (short)0, accS[ni], false, false);
        }
    }

    const float* rsb   = rs_b + layer * C2;
    const int rowsel   = (lane >> 4) << 3;
    #pragma unroll
    for (int ni = 0; ni < 4; ++ni) {
        const int col = ni * 16 + nc;
        #pragma unroll
        for (int v = 0; v < 8; ++v) {
            const int m = wv * 16 + rowsel + v;
            const size_t idx = (((size_t)b * CC + m) * HH + hh) * WW + w0 + col;
            float r  = accR[ni][v] + rsb[m];
            float hn = hf[idx] + r;
            hf[idx]  = hn;
            hbf[idx] = f2bf(hn);
            float s  = accS[ni][v] + rsb[128 + m];
            skip[idx] = first ? s : (skip[idx] + s);
        }
    }
}

// ---------------------------------- host -----------------------------------

extern "C" void kernel_launch(void* const* d_in, const int* in_sizes, int n_in,
                              void* d_out, int out_size, void* d_ws, size_t ws_size,
                              hipStream_t stream) {
    const float* x       = (const float*)d_in[0];
    const float* c       = (const float*)d_in[1];
    const float* front_w = (const float*)d_in[2];
    const float* front_b = (const float*)d_in[3];
    const float* fg_w    = (const float*)d_in[4];
    const float* fg_b    = (const float*)d_in[5];
    const float* fgc_w   = (const float*)d_in[6];
    const float* fgc_b   = (const float*)d_in[7];
    const float* rs_w    = (const float*)d_in[8];
    const float* rs_b    = (const float*)d_in[9];
    float* skip = (float*)d_out;

    // workspace layout (256B aligned regions)
    char* ws = (char*)d_ws;
    const size_t SZ_HF   = (size_t)BB * CC * HW * 4;   // 67 MB  h master f32
    const size_t SZ_HBF  = (size_t)BB * CC * HW * 2;   // 33 MB  h bf16 shadow
    const size_t SZ_ACT  = (size_t)BB * CC * HW * 2;   // 33 MB  acts bf16
    const size_t SZ_CBF  = (size_t)BB * 80 * HW * 2;   // 21 MB  c bf16
    const size_t SZ_WFG  = (size_t)NL * 9 * C2 * CC * 2;
    const size_t SZ_WFGC = (size_t)NL * C2 * CC * 2;
    float*          hf    = (float*)         (ws);
    unsigned short* hbf   = (unsigned short*)(ws + SZ_HF);
    unsigned short* actsb = (unsigned short*)(ws + SZ_HF + SZ_HBF);
    unsigned short* cbf   = (unsigned short*)(ws + SZ_HF + SZ_HBF + SZ_ACT);
    unsigned short* wfg   = (unsigned short*)(ws + SZ_HF + SZ_HBF + SZ_ACT + SZ_CBF);
    unsigned short* wfgc  = (unsigned short*)(ws + SZ_HF + SZ_HBF + SZ_ACT + SZ_CBF + SZ_WFG);
    unsigned short* wrs   = (unsigned short*)(ws + SZ_HF + SZ_HBF + SZ_ACT + SZ_CBF + SZ_WFG + SZ_WFGC);

    // precision conversions (deterministic, every call)
    k_cvt_fgw<<<(NL * 9 * C2 * CC + 255) / 256, 256, 0, stream>>>(fg_w, wfg);
    k_cvt_fgc<<<(NL * C2 * CC + 255) / 256, 256, 0, stream>>>(fgc_w, wfgc);
    k_cvt_rs <<<(NL * C2 * CC + 255) / 256, 256, 0, stream>>>(rs_w, wrs);
    k_cvt_c  <<<(BB * 80 * HW + 255) / 256, 256, 0, stream>>>(c, cbf);

    // front 1x1
    k_front<<<(BB * CC * HW + 255) / 256, 256, 0, stream>>>(x, front_w, front_b, hf, hbf);

    const int DH[NL] = {1, 2, 4, 8, 1, 2, 4, 8};
    const int DW[NL] = {1, 2, 4, 8, 16, 32, 64, 128};
    const int tiles  = NTOT / 64;   // 2048 workgroups, 8 waves each

    for (int l = 0; l < NL; ++l) {
        k_gate<<<tiles, 256, 0, stream>>>(hbf, cbf, wfg, wfgc, fg_b, fgc_b,
                                          actsb, l, DH[l], DW[l]);
        k_rs<<<tiles, 256, 0, stream>>>(actsb, wrs, rs_b, hf, hbf, skip,
                                        l, l == 0 ? 1 : 0);
    }
    (void)in_sizes; (void)n_in; (void)out_size; (void)ws_size;
}
